// SSMBlock_31937376813582
// MI455X (gfx1250) — matrix operand order repaired and checked by reading, compiled, not run on hardware
//
#include <hip/hip_runtime.h>
#include <hip/hip_bf16.h>
#include <cstdint>
#include <cstddef>

// ---------------------------------------------------------------------------
// Types / constants
// ---------------------------------------------------------------------------
typedef __bf16 bf16;
typedef __attribute__((ext_vector_type(16))) __bf16 bf16x16;
typedef __attribute__((ext_vector_type(8)))  __bf16 bf16x8;
typedef __attribute__((ext_vector_type(8)))  float  f32x8;
typedef __attribute__((ext_vector_type(4)))  int    i32x4;

constexpr int HIDDEN = 1024;
constexpr int STATE  = 16;
constexpr int BATCH  = 4;
constexpr int SEQ    = 2048;
constexpr int MROWS  = BATCH * SEQ;      // 8192 rows for all GEMMs
constexpr int NSEG   = 16;               // scan segments (parallel-scan chunks)
constexpr int LSEG   = SEQ / NSEG;       // 128 steps per segment
constexpr float LN_EPS = 1e-5f;

// GEMM tiling: block 128x128, K-step 32, 8 waves (wave tile 64x32 = 4x2 WMMA)
constexpr int BM = 128, BN = 128, BK = 32;
constexpr int ROWH = 40;                 // LDS row stride in bf16 (80B) -> conflict-free b128 reads

#define DEV static __device__ __forceinline__

// ---------------------------------------------------------------------------
// Optional gfx1250 async global->LDS path (compile-safe via __has_builtin)
// ---------------------------------------------------------------------------
#if defined(__gfx1250__) && __has_builtin(__builtin_amdgcn_global_load_async_to_lds_b128) && __has_builtin(__builtin_amdgcn_s_wait_asynccnt)
#define USE_ASYNC_LDS 1
#else
#define USE_ASYNC_LDS 0
#endif

// Copy a BM x BK bf16 tile (row-major, leading dim ldg) into padded LDS tile.
DEV void stage_tile(bf16* __restrict__ lds_tile, const bf16* __restrict__ g,
                    int ldg, int tid)
{
    const int c  = tid & 3;      // 16-byte chunk within the 64B row
    const int r0 = tid >> 2;     // 0..63
#pragma unroll
    for (int i = 0; i < 2; ++i) {
        const int r = r0 + i * 64;
        const bf16* src = g + (size_t)r * ldg + c * 8;
        bf16*       dst = lds_tile + r * ROWH + c * 8;
#if USE_ASYNC_LDS
        __builtin_amdgcn_global_load_async_to_lds_b128(
            (__attribute__((address_space(1))) i32x4*)(src),
            (__attribute__((address_space(3))) i32x4*)(dst), 0, 0);
#else
        *(bf16x8*)dst = *(const bf16x8*)src;
#endif
    }
}

DEV void wait_stage()
{
#if USE_ASYNC_LDS
    __builtin_amdgcn_s_wait_asynccnt(0);
#endif
    __syncthreads();
}

// 16-bit A 16x32 (K order matches B): lane<16: K elem {0..15};
// lane>=16: {16..31}.  (byte offsets {0,16} / {32,48})
DEV bf16x16 load_frag_a(const bf16* tile, int row0, int lane)
{
    const int r = row0 + (lane & 15);
    const int h = (lane >> 4) << 4;                  // 0 or 16 elems
    const bf16* p = tile + r * ROWH;
    bf16x8 lo = *(const bf16x8*)(p + h);
    bf16x8 hi = *(const bf16x8*)(p + h + 8);
    return __builtin_shufflevector(lo, hi, 0,1,2,3,4,5,6,7,8,9,10,11,12,13,14,15);
}

// 16-bit B 32x16: lane n<16 holds K=0..15 contiguous, lane n+16 holds K=16..31.
// B[k][n] = W[n][k]  ->  contiguous along K of row n of W.
DEV bf16x16 load_frag_b(const bf16* tile, int row0, int lane)
{
    const int r = row0 + (lane & 15);
    const int h = (lane >> 4) << 4;                  // 0 or 16 elems
    const bf16* p = tile + r * ROWH + h;
    bf16x8 lo = *(const bf16x8*)(p);
    bf16x8 hi = *(const bf16x8*)(p + 8);
    return __builtin_shufflevector(lo, hi, 0,1,2,3,4,5,6,7,8,9,10,11,12,13,14,15);
}

// ---------------------------------------------------------------------------
// f32 -> bf16 conversion (weights / activations for WMMA operands)
// ---------------------------------------------------------------------------
__global__ void cvt_bf16_kernel(const float* __restrict__ in, bf16* __restrict__ out, int n)
{
    int i = blockIdx.x * blockDim.x + threadIdx.x;
    const int stride = gridDim.x * blockDim.x;
    for (; i < n; i += stride) out[i] = (bf16)in[i];
}

// ---------------------------------------------------------------------------
// Generic bf16 WMMA GEMM:  out = A(MxK) * W(NxK)^T + bias, fused epilogues
// ---------------------------------------------------------------------------
enum { EPI_XZ = 0, EPI_DT = 1, EPI_OUT = 2 };

template <int EPI>
__global__ __launch_bounds__(256)
void gemm_bf16_kernel(const bf16* __restrict__ A, const bf16* __restrict__ W,
                      const float* __restrict__ bias, int K,
                      float* __restrict__ out_f,   // XZ: x_ssm f32 | DT: dt | OUT: pre
                      bf16*  __restrict__ out_bf,  // XZ: x_ssm bf16
                      float* __restrict__ out_f2,  // XZ: silu(z)
                      const float* __restrict__ resid) // OUT: residual x
{
    __shared__ __align__(16) bf16 lds[2][2][BM * ROWH];   // [buf][A|W]
    const int tid  = threadIdx.x;
    const int lane = tid & 31;
    const int wave = tid >> 5;
    const int wm = wave & 1;     // 2 wave rows  -> 64 rows each
    const int wn = wave >> 1;    // 4 wave cols  -> 32 cols each
    const int m0 = blockIdx.y * BM;
    const int n0 = blockIdx.x * BN;

    f32x8 acc[4][2] = {};
    const bf16* Ablk = A + (size_t)m0 * K;
    const bf16* Wblk = W + (size_t)n0 * K;

    int cur = 0;
    stage_tile(lds[0][0], Ablk, K, tid);
    stage_tile(lds[0][1], Wblk, K, tid);

    const int KT = K / BK;
    for (int kt = 0; kt < KT; ++kt) {
        wait_stage();                                   // buffer `cur` ready
        if (kt + 1 < KT) {                              // prefetch next K-slab
            stage_tile(lds[cur ^ 1][0], Ablk + (kt + 1) * BK, K, tid);
            stage_tile(lds[cur ^ 1][1], Wblk + (kt + 1) * BK, K, tid);
        }
        bf16x16 bfr[2];
#pragma unroll
        for (int tn = 0; tn < 2; ++tn)
            bfr[tn] = load_frag_b(lds[cur][1], wn * 32 + tn * 16, lane);
#pragma unroll
        for (int tm = 0; tm < 4; ++tm) {
            bf16x16 afr = load_frag_a(lds[cur][0], wm * 64 + tm * 16, lane);
#pragma unroll
            for (int tn = 0; tn < 2; ++tn)
                acc[tm][tn] = __builtin_amdgcn_wmma_f32_16x16x32_bf16(
                    false, afr, false, bfr[tn], (short)0, acc[tm][tn], false, false);
        }
        __syncthreads();                                // done reading `cur`
        cur ^= 1;
    }

    // Epilogue. C/D layout: VGPR r: lanes 0-15 -> (M=r, N=lane); 16-31 -> (M=8+r).
    const int nbase = n0 + wn * 32 + (lane & 15);
    const int mbase = m0 + wm * 64 + ((lane >> 4) << 3);
#pragma unroll
    for (int tn = 0; tn < 2; ++tn) {
        const int n  = nbase + tn * 16;
        const float bv = bias[n];
#pragma unroll
        for (int tm = 0; tm < 4; ++tm) {
            const int mrow = mbase + tm * 16;
#pragma unroll
            for (int r = 0; r < 8; ++r) {
                const int m = mrow + r;
                float v = acc[tm][tn][r] + bv;
                if constexpr (EPI == EPI_XZ) {
                    if (n < HIDDEN) {
                        const size_t off = (size_t)m * HIDDEN + n;
                        out_f[off]  = v;
                        out_bf[off] = (bf16)v;
                    } else {
                        const size_t off = (size_t)m * HIDDEN + (n - HIDDEN);
                        out_f2[off] = v / (1.f + __expf(-v));          // SiLU gate
                    }
                } else if constexpr (EPI == EPI_DT) {
                    const size_t off = (size_t)m * HIDDEN + n;
                    out_f[off] = (v > 20.f) ? v : log1pf(__expf(v));   // softplus
                } else {
                    const size_t off = (size_t)m * HIDDEN + n;
                    out_f[off] = v + resid[off];                       // + residual
                }
            }
        }
    }
}

// ---------------------------------------------------------------------------
// B/C projections: (8192 x 1024) x (16 x 1024)^T, one wave per 16-row tile.
// ---------------------------------------------------------------------------
__global__ __launch_bounds__(32)
void bc_gemm_kernel(const bf16* __restrict__ Xs, const bf16* __restrict__ wB,
                    const bf16* __restrict__ wC, const float* __restrict__ bB,
                    const float* __restrict__ bC,
                    float* __restrict__ Bm, float* __restrict__ Cm)
{
    const int lane = threadIdx.x & 31;
    const int m0   = blockIdx.x * 16;
    const int rr   = lane & 15;
    const int half = lane >> 4;
    f32x8 aB = {}, aC = {};
    const bf16* pa = Xs + (size_t)(m0 + rr) * HIDDEN;
    const bf16* pb = wB + (size_t)rr * HIDDEN;
    const bf16* pc = wC + (size_t)rr * HIDDEN;
    for (int k0 = 0; k0 < HIDDEN; k0 += 32) {
        bf16x8 alo = *(const bf16x8*)(pa + k0 + half * 16);
        bf16x8 ahi = *(const bf16x8*)(pa + k0 + half * 16 + 8);
        bf16x16 af = __builtin_shufflevector(alo, ahi, 0,1,2,3,4,5,6,7,8,9,10,11,12,13,14,15);
        bf16x8 blo = *(const bf16x8*)(pb + k0 + half * 16);
        bf16x8 bhi = *(const bf16x8*)(pb + k0 + half * 16 + 8);
        bf16x16 bfm = __builtin_shufflevector(blo, bhi, 0,1,2,3,4,5,6,7,8,9,10,11,12,13,14,15);
        bf16x8 clo = *(const bf16x8*)(pc + k0 + half * 16);
        bf16x8 chi = *(const bf16x8*)(pc + k0 + half * 16 + 8);
        bf16x16 cfm = __builtin_shufflevector(clo, chi, 0,1,2,3,4,5,6,7,8,9,10,11,12,13,14,15);
        aB = __builtin_amdgcn_wmma_f32_16x16x32_bf16(false, af, false, bfm, (short)0, aB, false, false);
        aC = __builtin_amdgcn_wmma_f32_16x16x32_bf16(false, af, false, cfm, (short)0, aC, false, false);
    }
    const int n  = lane & 15;
    const int mb = m0 + half * 8;
#pragma unroll
    for (int r = 0; r < 8; ++r) {
        Bm[(size_t)(mb + r) * STATE + n] = aB[r] + bB[n];
        Cm[(size_t)(mb + r) * STATE + n] = aC[r] + bC[n];
    }
}

// ---------------------------------------------------------------------------
// Chunked selective scan.
// Phase 1 (FINAL=false): per-segment h_end and decay product P = prod(exp(A*dt)).
// Phase 3 (FINAL=true):  replay with correct h0, fuse y = C.h + D*x, *silu(z),
//                        emit bf16 operand for the output GEMM.
// ---------------------------------------------------------------------------
template <bool FINAL>
__global__ __launch_bounds__(128)
void scan_seg_kernel(const float* __restrict__ dt, const float* __restrict__ xs,
                     const float* __restrict__ Bm, const float* __restrict__ Cm,
                     const float* __restrict__ A_log, const float* __restrict__ Dvec,
                     const float* __restrict__ z,
                     float* __restrict__ hend, float* __restrict__ Pseg,
                     const float* __restrict__ h0seg, bf16* __restrict__ yz_bf)
{
    const int d = blockIdx.x * 128 + threadIdx.x;
    const int s = blockIdx.y;
    const int b = blockIdx.z;

    float a[STATE];
#pragma unroll
    for (int n = 0; n < STATE; ++n) a[n] = -__expf(A_log[d * STATE + n]);

    const size_t segbase = ((size_t)(b * NSEG + s) * HIDDEN + d) * STATE;
    float h[STATE], P[STATE];
#pragma unroll
    for (int n = 0; n < STATE; ++n) {
        h[n] = FINAL ? h0seg[segbase + n] : 0.f;
        P[n] = 1.f;
    }
    const float dval = FINAL ? Dvec[d] : 0.f;

    __shared__ float sB[32][STATE];
    __shared__ float sC[32][STATE];
    const int t0 = s * LSEG;

    for (int tc = 0; tc < LSEG; tc += 32) {
        // cooperative load of 32 steps of B (and C) into LDS
        for (int i = threadIdx.x; i < 32 * STATE; i += 128) {
            const int tt = i >> 4, nn = i & 15;
            const size_t bo = (size_t)(b * SEQ + t0 + tc + tt) * STATE + nn;
            sB[tt][nn] = Bm[bo];
            if (FINAL) sC[tt][nn] = Cm[bo];
        }
        __syncthreads();

        for (int j = 0; j < 32; ++j) {
            const int t = t0 + tc + j;
            const size_t off = (size_t)(b * SEQ + t) * HIDDEN + d;
            const float dtv = dt[off];
            const float xv  = xs[off];
            const float dtx = dtv * xv;
            float y = 0.f;
#pragma unroll
            for (int n = 0; n < STATE; ++n) {
                const float dA = __expf(a[n] * dtv);
                h[n] = fmaf(dA, h[n], sB[j][n] * dtx);
                if constexpr (!FINAL) P[n] *= dA;
                if constexpr (FINAL)  y = fmaf(h[n], sC[j][n], y);
            }
            if constexpr (FINAL) {
                y = fmaf(dval, xv, y);          // + D * x_ssm
                yz_bf[off] = (bf16)(y * z[off]); // * silu(z), bf16 for out-GEMM
            }
        }
        __syncthreads();
    }
    if constexpr (!FINAL) {
#pragma unroll
        for (int n = 0; n < STATE; ++n) {
            hend[segbase + n] = h[n];
            Pseg[segbase + n] = P[n];
        }
    }
}

// Phase 2: tiny sequential combine across segments, one thread per (b,d,n).
__global__ __launch_bounds__(256)
void scan_combine_kernel(const float* __restrict__ hend, const float* __restrict__ Pseg,
                         float* __restrict__ h0seg)
{
    const int i = blockIdx.x * 256 + threadIdx.x;     // over b * (D*N)
    if (i >= BATCH * HIDDEN * STATE) return;
    const int b   = i / (HIDDEN * STATE);
    const int rem = i % (HIDDEN * STATE);
    float carry = 0.f;
#pragma unroll
    for (int s = 0; s < NSEG; ++s) {
        const size_t base = (size_t)(b * NSEG + s) * HIDDEN * STATE + rem;
        h0seg[base] = carry;
        carry = fmaf(Pseg[base], carry, hend[base]);
    }
}

// ---------------------------------------------------------------------------
// LayerNorm: one block per row of `pre`
// ---------------------------------------------------------------------------
__global__ __launch_bounds__(256)
void layernorm_kernel(const float* __restrict__ pre, const float* __restrict__ gamma,
                      const float* __restrict__ beta, float* __restrict__ out)
{
    const int row = blockIdx.x;
    const int tid = threadIdx.x;
    const float* p = pre + (size_t)row * HIDDEN;
    float v[4], s = 0.f, s2 = 0.f;
#pragma unroll
    for (int i = 0; i < 4; ++i) {
        v[i] = p[tid * 4 + i];
        s += v[i];
        s2 += v[i] * v[i];
    }
#pragma unroll
    for (int o = 16; o > 0; o >>= 1) {
        s  += __shfl_xor(s,  o, 32);
        s2 += __shfl_xor(s2, o, 32);
    }
    __shared__ float r1[8], r2[8];
    if ((tid & 31) == 0) { r1[tid >> 5] = s; r2[tid >> 5] = s2; }
    __syncthreads();
    s = 0.f; s2 = 0.f;
#pragma unroll
    for (int w = 0; w < 8; ++w) { s += r1[w]; s2 += r2[w]; }
    const float mu  = s * (1.f / HIDDEN);
    const float var = s2 * (1.f / HIDDEN) - mu * mu;
    const float inv = rsqrtf(var + LN_EPS);
#pragma unroll
    for (int i = 0; i < 4; ++i) {
        const int c = tid * 4 + i;
        out[(size_t)row * HIDDEN + c] = (v[i] - mu) * inv * gamma[c] + beta[c];
    }
}

// ---------------------------------------------------------------------------
// Host launcher
// ---------------------------------------------------------------------------
static inline void cvt(const float* in, bf16* out, int n, hipStream_t s)
{
    int blocks = (n + 255) / 256;
    if (blocks > 2048) blocks = 2048;
    cvt_bf16_kernel<<<blocks, 256, 0, s>>>(in, out, n);
}

extern "C" void kernel_launch(void* const* d_in, const int* in_sizes, int n_in,
                              void* d_out, int out_size, void* d_ws, size_t ws_size,
                              hipStream_t stream)
{
    (void)in_sizes; (void)n_in; (void)out_size; (void)ws_size;
    const float* x     = (const float*)d_in[0];
    const float* w_in  = (const float*)d_in[1];
    const float* b_in  = (const float*)d_in[2];
    const float* w_dt  = (const float*)d_in[3];
    const float* b_dt  = (const float*)d_in[4];
    const float* A_log = (const float*)d_in[5];
    const float* w_B   = (const float*)d_in[6];
    const float* b_B   = (const float*)d_in[7];
    const float* w_C   = (const float*)d_in[8];
    const float* b_C   = (const float*)d_in[9];
    const float* Dv    = (const float*)d_in[10];
    const float* w_out = (const float*)d_in[11];
    const float* b_out = (const float*)d_in[12];
    const float* gamma = (const float*)d_in[13];
    const float* beta  = (const float*)d_in[14];

    char* ws = (char*)d_ws;
    size_t off = 0;
    auto alloc = [&](size_t bytes) -> char* {
        char* p = ws + off;
        off += (bytes + 255) & ~(size_t)255;
        return p;
    };
    const size_t MD = (size_t)MROWS * HIDDEN;
    float* xssm_f  = (float*)alloc(MD * 4);
    float* z_f     = (float*)alloc(MD * 4);
    float* dt_f    = (float*)alloc(MD * 4);
    float* Bm      = (float*)alloc((size_t)MROWS * STATE * 4);
    float* Cm      = (float*)alloc((size_t)MROWS * STATE * 4);
    float* hend    = (float*)alloc((size_t)BATCH * NSEG * HIDDEN * STATE * 4);
    float* Pseg    = (float*)alloc((size_t)BATCH * NSEG * HIDDEN * STATE * 4);
    float* h0seg   = (float*)alloc((size_t)BATCH * NSEG * HIDDEN * STATE * 4);
    float* pre     = (float*)alloc(MD * 4);
    bf16*  x_bf    = (bf16*)alloc(MD * 2);
    bf16*  xssm_bf = (bf16*)alloc(MD * 2);
    bf16*  yz_bf   = (bf16*)alloc(MD * 2);
    bf16*  w_in_bf = (bf16*)alloc((size_t)2 * HIDDEN * HIDDEN * 2);
    bf16*  w_dt_bf = (bf16*)alloc((size_t)HIDDEN * HIDDEN * 2);
    bf16*  w_out_bf= (bf16*)alloc((size_t)HIDDEN * HIDDEN * 2);
    bf16*  wB_bf   = (bf16*)alloc((size_t)STATE * HIDDEN * 2);
    bf16*  wC_bf   = (bf16*)alloc((size_t)STATE * HIDDEN * 2);

    // 0) bf16 copies of WMMA operands
    cvt(x,     x_bf,     (int)MD, stream);
    cvt(w_in,  w_in_bf,  2 * HIDDEN * HIDDEN, stream);
    cvt(w_dt,  w_dt_bf,  HIDDEN * HIDDEN, stream);
    cvt(w_out, w_out_bf, HIDDEN * HIDDEN, stream);
    cvt(w_B,   wB_bf,    STATE * HIDDEN, stream);
    cvt(w_C,   wC_bf,    STATE * HIDDEN, stream);

    // 1) xz = x @ w_in^T + b_in  ->  x_ssm (f32 + bf16), silu(z)
    gemm_bf16_kernel<EPI_XZ><<<dim3(2 * HIDDEN / BN, MROWS / BM), 256, 0, stream>>>(
        x_bf, w_in_bf, b_in, HIDDEN, xssm_f, xssm_bf, z_f, nullptr);

    // 2) dt = softplus(x_ssm @ w_dt^T + b_dt)
    gemm_bf16_kernel<EPI_DT><<<dim3(HIDDEN / BN, MROWS / BM), 256, 0, stream>>>(
        xssm_bf, w_dt_bf, b_dt, HIDDEN, dt_f, nullptr, nullptr, nullptr);

    // 3) Bm / Cm projections (N=16 each), one wave per 16-row tile
    bc_gemm_kernel<<<MROWS / 16, 32, 0, stream>>>(xssm_bf, wB_bf, wC_bf, b_B, b_C, Bm, Cm);

    // 4) chunked selective scan: partial -> combine -> final (fused gate + D*x)
    dim3 sgrid(HIDDEN / 128, NSEG, BATCH);
    scan_seg_kernel<false><<<sgrid, 128, 0, stream>>>(
        dt_f, xssm_f, Bm, Cm, A_log, Dv, z_f, hend, Pseg, nullptr, nullptr);
    scan_combine_kernel<<<(BATCH * HIDDEN * STATE + 255) / 256, 256, 0, stream>>>(
        hend, Pseg, h0seg);
    scan_seg_kernel<true><<<sgrid, 128, 0, stream>>>(
        dt_f, xssm_f, Bm, Cm, A_log, Dv, z_f, nullptr, nullptr, h0seg, yz_bf);

    // 5) pre = (y*z) @ w_out^T + b_out + x
    gemm_bf16_kernel<EPI_OUT><<<dim3(HIDDEN / BN, MROWS / BM), 256, 0, stream>>>(
        yz_bf, w_out_bf, b_out, HIDDEN, pre, nullptr, nullptr, x);

    // 6) LayerNorm
    layernorm_kernel<<<MROWS, 256, 0, stream>>>(pre, gamma, beta, (float*)d_out);
}
